// LightNetAttention_12738873000587
// MI455X (gfx1250) — matrix operand order repaired and checked by reading, not compiled, not run on hardware
//
#include <hip/hip_runtime.h>
#include <hip/hip_bf16.h>
#include <stdint.h>

// ---------------- problem constants (from reference) ----------------
#define H_HEADS   16
#define HD        64
#define DMODEL    1024
#define BATCH     2
#define SEQ       2048
#define MROWS     (BATCH * SEQ)      // 4096 rows in all GEMMs
#define LRPE_BASE 10000.0f
#define RMS_EPS   1e-6f

// ---------------- vector types ----------------
typedef __attribute__((ext_vector_type(8)))  float          v8f;
typedef __attribute__((ext_vector_type(16))) __bf16         bf16x16;
typedef __attribute__((ext_vector_type(16), aligned(16))) unsigned short u16x16a;

static __device__ __forceinline__ uint16_t f2bf(float f) {
  uint32_t u = __float_as_uint(f);
  uint32_t r = u + 0x7FFFu + ((u >> 16) & 1u);   // round-to-nearest-even
  return (uint16_t)(r >> 16);
}

// 16-byte async copy global -> LDS (tracked by ASYNCcnt)
static __device__ __forceinline__ void async_copy16(uint32_t lds_off, const void* gptr) {
  asm volatile("global_load_async_to_lds_b128 %0, %1, off"
               :: "v"(lds_off), "v"((unsigned long long)(uintptr_t)gptr)
               : "memory");
}
static __device__ __forceinline__ void wait_async() {
  asm volatile("s_wait_asynccnt 0x0" ::: "memory");
}

// ---------------- fp32 -> bf16 conversion ----------------
__global__ void cvt_f32_to_bf16(const float* __restrict__ in,
                                uint16_t* __restrict__ out, int nelem) {
  int i = blockIdx.x * blockDim.x + threadIdx.x;
  int stride = gridDim.x * blockDim.x;
  for (; i < nelem; i += stride) out[i] = f2bf(in[i]);
}

// fp32 (RxC row-major) -> bf16 transposed (CxR row-major)
__global__ void cvt_f32_to_bf16_T(const float* __restrict__ in,
                                  uint16_t* __restrict__ out, int R, int C) {
  int i = blockIdx.x * blockDim.x + threadIdx.x;
  int stride = gridDim.x * blockDim.x;
  int total = R * C;
  for (; i < total; i += stride) {
    int r = i / C, c = i % C;
    out[(size_t)c * R + r] = f2bf(in[i]);
  }
}

// ---------------- WMMA bf16 GEMM: C(MxN,f32) = A(MxK,bf16) * Bt(NxK,bf16)^T --
// B operand stored TRANSPOSED (N-major) so both fragments load as 2x b128.
// Double-buffered LDS; async global->LDS staging of tile kt+1 overlaps the
// WMMA compute of tile kt. One barrier per K-step.
// Block: 256 threads = 8 waves. Tile BM=128, BN=128, BK=32.
enum { EP_NONE_F32 = 0, EP_SILU_F32 = 1, EP_NONE_BF16 = 2 };

template <int EP, bool FULLN>
__global__ __launch_bounds__(256) void gemm_bf16_wmma(
    const uint16_t* __restrict__ A, const uint16_t* __restrict__ Bt,
    void* __restrict__ Cv, int M, int N, int K) {
  (void)M;
  // A tile stored with K-chunk permutation [0,2,1,3] so each lane's 16
  // fragment halves (K = lh*8+0..7 ++ 16+lh*8+0..7) are contiguous.
  __shared__ uint16_t As[2][128][40];   // 2 x (128 x 32 bf16, +8 pad)
  __shared__ uint16_t Bs[2][128][40];   // 2 x (128(n) x 32(k) bf16)

  const int t    = threadIdx.x;
  const int lane = t & 31;
  const int wid  = t >> 5;
  const int lr   = lane & 15;   // row (A) / col (B,C)
  const int lh   = lane >> 4;   // half selector
  const int wr   = wid >> 1;    // 0..3 -> 32-row strip
  const int wc   = wid & 1;     // 0..1 -> 64-col strip
  const int m0   = blockIdx.y * 128;
  const int n0   = blockIdx.x * 128;

  v8f acc[2][4] = {};
  const int nk = K >> 5;

  // stage one 128x32 A tile + 128x32 B tile into buffer `buf` (async)
  auto stage = [&](int kt, int buf) {
    const int k0 = kt << 5;
#pragma unroll
    for (int i = 0; i < 2; ++i) {
      int idx = t + (i << 8);       // 0..511
      int row = idx >> 2;           // 0..127
      int cc  = idx & 3;            // 8-half K chunk
      int pos = ((cc & 1) << 1) | (cc >> 1);  // 0,1,2,3 -> 0,2,1,3
      async_copy16((uint32_t)(uintptr_t)&As[buf][row][pos * 8],
                   A + (size_t)(m0 + row) * K + (k0 + cc * 8));
      if (FULLN || (n0 + row < N)) {
        async_copy16((uint32_t)(uintptr_t)&Bs[buf][row][pos * 8],
                     Bt + (size_t)(n0 + row) * K + (k0 + cc * 8));
      } else {
        *(uint4*)&Bs[buf][row][cc * 8] = make_uint4(0u, 0u, 0u, 0u);
      }
    }
  };

  stage(0, 0);
  wait_async();
  __syncthreads();

  for (int kt = 0; kt < nk; ++kt) {
    const int cur = kt & 1;
    const int nxt = cur ^ 1;
    // kick off next tile's staging before computing this one (latency overlap)
    if (kt + 1 < nk) stage(kt + 1, nxt);
    if (kt + 2 < nk) {  // warm L2 two tiles ahead (global_prefetch_b8)
      __builtin_prefetch(A + (size_t)(m0 + (t >> 2)) * K + ((kt + 2) << 5), 0, 0);
      __builtin_prefetch(Bt + (size_t)(n0 + (t >> 2)) * K + ((kt + 2) << 5), 0, 0);
    }

    // ---- A fragments: contiguous 16 halves per lane -> 2x ds_load_b128 ----
    bf16x16 afrag[2];
#pragma unroll
    for (int mt = 0; mt < 2; ++mt) {
      int r = wr * 32 + mt * 16 + lr;
      u16x16a w = *(const u16x16a*)&As[cur][r][lh * 16];
      afrag[mt] = __builtin_bit_cast(bf16x16, w);
    }
    // ---- B fragments: Bs K-chunks permuted [0,2,1,3] like As; halves = K lh*8+0..7 ++ 16+lh*8+0..7 ----
#pragma unroll
    for (int nt = 0; nt < 4; ++nt) {
      int cb = wc * 64 + nt * 16 + lr;   // column (n) within block tile
      u16x16a w = *(const u16x16a*)&Bs[cur][cb][lh * 16];
      bf16x16 bfrag = __builtin_bit_cast(bf16x16, w);
#pragma unroll
      for (int mt = 0; mt < 2; ++mt) {
        acc[mt][nt] = __builtin_amdgcn_wmma_f32_16x16x32_bf16(
            false, afrag[mt], false, bfrag, (short)0, acc[mt][nt], false, false);
      }
    }
    // next buffer's async loads must land, and everyone must be done reading
    // `cur`, before the next iteration overwrites buffers
    wait_async();
    __syncthreads();
  }

  // ---- epilogue: C/D layout — VGPR r: lanes 0-15 -> M=r, lanes 16-31 -> M=r+8
  float*    Cf = (float*)Cv;
  uint16_t* Cb = (uint16_t*)Cv;
#pragma unroll
  for (int mt = 0; mt < 2; ++mt) {
#pragma unroll
    for (int nt = 0; nt < 4; ++nt) {
#pragma unroll
      for (int r = 0; r < 8; ++r) {
        int row = m0 + wr * 32 + mt * 16 + lh * 8 + r;
        int col = n0 + wc * 64 + nt * 16 + lr;
        if (FULLN || col < N) {
          float val = acc[mt][nt][r];
          if (EP == EP_SILU_F32) val = val / (1.0f + __expf(-val));  // x*sigmoid(x)
          if (EP == EP_NONE_BF16) Cb[(size_t)row * N + col] = f2bf(val);
          else                    Cf[(size_t)row * N + col] = val;
        }
      }
    }
  }
}

// ---------------- streaming cumlogsumexp over time (z seeded by 0-row) --------
__global__ void cumlse_kernel(const float* __restrict__ kproj,
                              float* __restrict__ logf_out,
                              float* __restrict__ kn_out) {
  int ch = blockIdx.x * blockDim.x + threadIdx.x;
  if (ch >= BATCH * DMODEL) return;
  int bi = ch / DMODEL;
  int c  = ch % DMODEL;
  const size_t base = (size_t)bi * SEQ * DMODEL + c;
  float z = 0.0f;
  for (int tt = 0; tt < SEQ; ++tt) {
    size_t idx = base + (size_t)tt * DMODEL;
    float kv   = kproj[idx];
    float m    = fmaxf(z, kv);
    float znew = m + __logf(__expf(z - m) + __expf(kv - m));
    logf_out[idx] = z - znew;           // log decay (<= 0)
    kn_out[idx]   = __expf(kv - znew);  // normalized exp kernel key
    z = znew;
  }
}

// ---------------- linear-attention recurrence with vector decay ---------------
// One block per (b,h); 64 threads; thread v owns S[:,v] (128 f32 in registers).
__global__ __launch_bounds__(64) void scan_kernel(
    const float* __restrict__ q, const float* __restrict__ kn,
    const float* __restrict__ v, const float* __restrict__ lf,
    float* __restrict__ o) {
  __shared__ float shq[128], shk[128], shd[128];
  const int bh   = blockIdx.x;
  const int bi   = bh >> 4;
  const int h    = bh & 15;
  const int vcol = threadIdx.x;   // 0..63
  const size_t base  = (size_t)bi * SEQ * DMODEL + (size_t)h * HD;
  const float  theta = __expf((-2.0f * __logf(LRPE_BASE) / (float)DMODEL) *
                              (float)(h * HD + vcol));
  float S[128];
#pragma unroll
  for (int e = 0; e < 128; ++e) S[e] = 0.0f;

  for (int tt = 0; tt < SEQ; ++tt) {
    size_t idx = base + (size_t)tt * DMODEL + vcol;
    float qv = q[idx], kv = kn[idx], vv = v[idx], lv = lf[idx];
    float sn, cs;
    __sincosf((float)tt * theta, &sn, &cs);
    shq[vcol] = qv * cs;  shq[vcol + 64] = qv * sn;
    shk[vcol] = kv * cs;  shk[vcol + 64] = kv * sn;
    float dec = __expf(lv);
    shd[vcol] = dec;      shd[vcol + 64] = dec;   // ld repeated d -> 2d
    __syncthreads();
    float ov = 0.0f;
#pragma unroll
    for (int e = 0; e < 128; ++e) {
      float s = fmaf(shd[e], S[e], shk[e] * vv);  // S = diag(exp(ld))S + k v^T
      S[e] = s;
      ov = fmaf(shq[e], s, ov);                   // o_v = sum_e q2_e * S[e][v]
    }
    o[idx] = ov;
    __syncthreads();
  }
}

// ---------------- fused sigmoid gate + group RMSNorm (groups = heads) ---------
__global__ __launch_bounds__(256) void gate_norm_kernel(
    const float* __restrict__ o, const float* __restrict__ gate,
    const float* __restrict__ nw, uint16_t* __restrict__ outbf) {
  const int lane = threadIdx.x & 31;
  const int wid  = threadIdx.x >> 5;
  const int gid  = blockIdx.x * 8 + wid;        // group index over b*n*H
  const size_t base = (size_t)gid * HD;
  float x0 = o[base + lane],      a0 = gate[base + lane];
  float x1 = o[base + lane + 32], a1 = gate[base + lane + 32];
  float g0 = x0 / (1.0f + __expf(-a0));
  float g1 = x1 / (1.0f + __expf(-a1));
  float ss = g0 * g0 + g1 * g1;
#pragma unroll
  for (int off = 16; off >= 1; off >>= 1) ss += __shfl_xor(ss, off, 32);
  float inv  = rsqrtf(ss * (1.0f / (float)HD) + RMS_EPS);
  int   col0 = (int)((base + lane) % DMODEL);
  int   col1 = (int)((base + lane + 32) % DMODEL);
  outbf[base + lane]      = f2bf(g0 * inv * nw[col0]);
  outbf[base + lane + 32] = f2bf(g1 * inv * nw[col1]);
}

// ---------------- launcher ----------------
extern "C" void kernel_launch(void* const* d_in, const int* in_sizes, int n_in,
                              void* d_out, int out_size, void* d_ws, size_t ws_size,
                              hipStream_t stream) {
  (void)in_sizes; (void)n_in; (void)out_size; (void)ws_size;
  const float* x   = (const float*)d_in[0];
  const float* wq  = (const float*)d_in[1];
  const float* wk  = (const float*)d_in[2];
  const float* wv  = (const float*)d_in[3];
  const float* wo  = (const float*)d_in[4];
  const float* wg1 = (const float*)d_in[5];
  const float* wg2 = (const float*)d_in[6];
  const float* nw  = (const float*)d_in[7];

  char*  ws  = (char*)d_ws;
  size_t off = 0;
  auto alloc = [&](size_t bytes) -> void* {
    off = (off + 255) & ~(size_t)255;
    void* p = ws + off;
    off += bytes;
    return p;
  };
  const size_t MD = (size_t)MROWS * DMODEL;
  uint16_t* xbf   = (uint16_t*)alloc(MD * 2);
  uint16_t* wqT   = (uint16_t*)alloc((size_t)DMODEL * DMODEL * 2);
  uint16_t* wkT   = (uint16_t*)alloc((size_t)DMODEL * DMODEL * 2);
  uint16_t* wvT   = (uint16_t*)alloc((size_t)DMODEL * DMODEL * 2);
  uint16_t* woT   = (uint16_t*)alloc((size_t)DMODEL * DMODEL * 2);
  uint16_t* wg1T  = (uint16_t*)alloc((size_t)DMODEL * HD * 2);
  uint16_t* wg2T  = (uint16_t*)alloc((size_t)HD * DMODEL * 2);
  float*    qf    = (float*)alloc(MD * 4);
  float*    kf    = (float*)alloc(MD * 4);
  float*    vf    = (float*)alloc(MD * 4);
  float*    lff   = (float*)alloc(MD * 4);
  float*    knf   = (float*)alloc(MD * 4);
  float*    gatef = (float*)alloc(MD * 4);
  float*    of    = (float*)alloc(MD * 4);
  uint16_t* g1b   = (uint16_t*)alloc((size_t)MROWS * HD * 2);
  uint16_t* nrmb  = (uint16_t*)alloc(MD * 2);

  // fp32 -> bf16; weights transposed (GEMM consumes Bt = N x K)
  cvt_f32_to_bf16  <<<512, 256, 0, stream>>>(x,   xbf,  (int)MD);
  cvt_f32_to_bf16_T<<<256, 256, 0, stream>>>(wq,  wqT,  DMODEL, DMODEL);
  cvt_f32_to_bf16_T<<<256, 256, 0, stream>>>(wk,  wkT,  DMODEL, DMODEL);
  cvt_f32_to_bf16_T<<<256, 256, 0, stream>>>(wv,  wvT,  DMODEL, DMODEL);
  cvt_f32_to_bf16_T<<<256, 256, 0, stream>>>(wo,  woT,  DMODEL, DMODEL);
  cvt_f32_to_bf16_T<<<64,  256, 0, stream>>>(wg1, wg1T, DMODEL, HD);
  cvt_f32_to_bf16_T<<<64,  256, 0, stream>>>(wg2, wg2T, HD, DMODEL);

  dim3 blk(256);
  dim3 gFull(DMODEL / 128, MROWS / 128);  // (8, 32)
  dim3 gNarrow(1, MROWS / 128);           // N = 64
  // projections + gate path (all WMMA)
  gemm_bf16_wmma<EP_SILU_F32, true><<<gFull, blk, 0, stream>>>(
      xbf, wqT, qf, MROWS, DMODEL, DMODEL);
  gemm_bf16_wmma<EP_NONE_F32, true><<<gFull, blk, 0, stream>>>(
      xbf, wkT, kf, MROWS, DMODEL, DMODEL);
  gemm_bf16_wmma<EP_NONE_F32, true><<<gFull, blk, 0, stream>>>(
      xbf, wvT, vf, MROWS, DMODEL, DMODEL);
  gemm_bf16_wmma<EP_NONE_BF16, false><<<gNarrow, blk, 0, stream>>>(
      xbf, wg1T, g1b, MROWS, HD, DMODEL);
  gemm_bf16_wmma<EP_NONE_F32, true><<<gFull, blk, 0, stream>>>(
      g1b, wg2T, gatef, MROWS, DMODEL, HD);
  // sequential parts
  cumlse_kernel<<<(BATCH * DMODEL + 255) / 256, 256, 0, stream>>>(kf, lff, knf);
  scan_kernel<<<BATCH * H_HEADS, 64, 0, stream>>>(qf, knf, vf, lff, of);
  gate_norm_kernel<<<(MROWS * H_HEADS) / 8, 256, 0, stream>>>(of, gatef, nw, nrmb);
  // output projection (WMMA) -> d_out fp32
  gemm_bf16_wmma<EP_NONE_F32, true><<<gFull, blk, 0, stream>>>(
      nrmb, woT, (float*)d_out, MROWS, DMODEL, DMODEL);
}
